// Block_9294309228733
// MI455X (gfx1250) — matrix operand order repaired and checked by reading, not compiled, not run on hardware
//
#include <hip/hip_runtime.h>
#include <cstdint>
#include <cstddef>

// ---------------- problem constants ----------------
constexpr int Bc  = 2;
constexpr int Tc  = 2048;
constexpr int Dc  = 1024;
constexpr int NHc = 16;
constexpr int DKc = 64;
constexpr int Lc  = 512;
constexpr int Ec  = 8;
constexpr int Kc  = 2;
constexpr int FFc = 4096;
constexpr int Mtok  = Bc * Tc;        // 4096 token rows
constexpr int NSLOT = Mtok * Kc;      // 8192 expert slots total (exact)
constexpr float NEGF = -9.0e15f;

// ---------------- CDNA5 WMMA types ----------------
typedef __bf16 bf16;
typedef __attribute__((ext_vector_type(16))) __bf16 v16bf;
typedef __attribute__((ext_vector_type(8)))  __bf16 v8bf;
typedef __attribute__((ext_vector_type(4)))  __bf16 v4bf;
typedef __attribute__((ext_vector_type(2)))  __bf16 v2bf;
typedef __attribute__((ext_vector_type(8)))  float  v8f;

// ---------------- GEMM tiling ----------------
constexpr int BMt = 128, BNt = 128, BKt = 32;
constexpr int APAD = 8;                 // halves; keeps rows 16B aligned, spreads banks
constexpr int AKs  = BKt + APAD;        // 40 halves = 80 bytes/row

__device__ __forceinline__ v16bf mk16(v8bf lo, v8bf hi) {
  v16bf r;
#pragma unroll
  for (int i = 0; i < 8; ++i) { r[i] = lo[i]; r[i + 8] = hi[i]; }
  return r;
}

__device__ __forceinline__ v4bf pack4(float a, float b, float c, float d) {
  v4bf r; r[0] = (bf16)a; r[1] = (bf16)b; r[2] = (bf16)c; r[3] = (bf16)d;
  return r;
}

// A fragment (16x32 bf16): lane l: m=l%16, g=l/16; elems 0..7 -> K=g*8+e, elems 8..15 -> K=16+g*8+(e-8)
__device__ __forceinline__ v16bf frag_a(const bf16* base, int row, int stride, int k0, int g) {
  const bf16* p = base + row * stride + k0;
  v8bf lo = *(const v8bf*)(p + g * 8);
  v8bf hi = *(const v8bf*)(p + 16 + g * 8);
  return mk16(lo, hi);
}

// B fragment (32x16 bf16), LDS stored transposed [n][k]: lane l: n=l%16, g=l/16; elems 0..7 -> K=g*8+e, elems 8..15 -> K=16+g*8+(e-8)
__device__ __forceinline__ v16bf frag_b(const bf16* base, int row, int stride, int k0, int g) {
  const bf16* p = base + row * stride + k0;
  v8bf lo = *(const v8bf*)(p + g * 8);
  v8bf hi = *(const v8bf*)(p + 16 + g * 8);
  return mk16(lo, hi);
}

// ---------------- shared GEMM core: C = op(A) @ W + bias ----------------
// A: fp32 [.. x K] rows addressed as gatherRows ? gatherRows[m] : aBase+m  (lda multiple of 4)
// W: fp32 [K x N] row-major (N multiple of 4).  C row = cBase + m.
// fp32 -> bf16 LDS staging with register double-buffering; fp32 WMMA accumulation.
__device__ __forceinline__ void gemm_core(
    const float* __restrict__ A, int lda, const int* __restrict__ gatherRows, int aBase,
    const float* __restrict__ W, const float* __restrict__ bias, int relu,
    float* __restrict__ C, int ldc, int cBase,
    int Mv, int N, int K, int tm0, int tn0)
{
  __shared__ bf16 Als[BMt * AKs];
  __shared__ bf16 Bls[BNt * AKs];

  const int tid  = threadIdx.x;     // 256 threads = 8 waves
  const int lane = tid & 31;
  const int wave = tid >> 5;
  const int g    = lane >> 4;
  const int l16  = lane & 15;
  const int wm   = wave & 3;        // 4 waves along M (32 rows each)
  const int wn   = wave >> 2;       // 2 waves along N (64 cols each)

  v8f acc[2][4];
#pragma unroll
  for (int si = 0; si < 2; ++si)
#pragma unroll
    for (int sj = 0; sj < 4; ++sj)
#pragma unroll
      for (int r = 0; r < 8; ++r) acc[si][sj][r] = 0.0f;

  // staging registers (prefetch buffer): A = 4 x float4, B = 4 x 4 floats
  float4 aR[4];
  float  bR[4][4];

  auto load_tiles = [&](int k0) {
#pragma unroll
    for (int u = 0; u < 4; ++u) {            // A: 128x32 as float4 chunks along k
      int i  = tid + u * 256;
      int m  = i >> 3;
      int kc = (i & 7) * 4;
      int gm = tm0 + m;
      float4 v = make_float4(0.0f, 0.0f, 0.0f, 0.0f);
      if (gm < Mv) {
        int ar = gatherRows ? gatherRows[gm] : (aBase + gm);
        v = *(const float4*)&A[(size_t)ar * lda + (size_t)(k0 + kc)];
      }
      aR[u] = v;
    }
#pragma unroll
    for (int u = 0; u < 4; ++u) {            // B: 32x128, 4 k's per n (coalesced in n)
      int i   = tid + u * 256;
      int n   = i & 127;
      int kb2 = (i >> 7) * 4;
      int gn  = tn0 + n;
#pragma unroll
      for (int j = 0; j < 4; ++j)
        bR[u][j] = (gn < N) ? W[(size_t)(k0 + kb2 + j) * N + gn] : 0.0f;
    }
  };

  auto store_tiles = [&]() {
#pragma unroll
    for (int u = 0; u < 4; ++u) {
      int i  = tid + u * 256;
      int m  = i >> 3;
      int kc = (i & 7) * 4;
      *(v4bf*)&Als[m * AKs + kc] = pack4(aR[u].x, aR[u].y, aR[u].z, aR[u].w);
    }
#pragma unroll
    for (int u = 0; u < 4; ++u) {
      int i   = tid + u * 256;
      int n   = i & 127;
      int kb2 = (i >> 7) * 4;
      *(v4bf*)&Bls[n * AKs + kb2] = pack4(bR[u][0], bR[u][1], bR[u][2], bR[u][3]);
    }
  };

  load_tiles(0);
  for (int k0 = 0; k0 < K; k0 += BKt) {
    store_tiles();
    __syncthreads();
    if (k0 + BKt < K) load_tiles(k0 + BKt);   // overlap next tile's HBM latency with WMMA

    v16bf bfr[4];
#pragma unroll
    for (int sj = 0; sj < 4; ++sj)
      bfr[sj] = frag_b(Bls, wn * 64 + sj * 16 + l16, AKs, 0, g);
#pragma unroll
    for (int si = 0; si < 2; ++si) {
      v16bf afr = frag_a(Als, wm * 32 + si * 16 + l16, AKs, 0, g);
#pragma unroll
      for (int sj = 0; sj < 4; ++sj) {
        acc[si][sj] = __builtin_amdgcn_wmma_f32_16x16x32_bf16(
            false, afr, false, bfr[sj], (short)0, acc[si][sj], false, false);
      }
    }
    __syncthreads();
  }

  // epilogue: C/D layout: lane col = l16, rows = g*8 + r
#pragma unroll
  for (int si = 0; si < 2; ++si)
#pragma unroll
    for (int sj = 0; sj < 4; ++sj)
#pragma unroll
      for (int r = 0; r < 8; ++r) {
        int row = tm0 + wm * 32 + si * 16 + g * 8 + r;
        int col = tn0 + wn * 64 + sj * 16 + l16;
        if (row < Mv && col < N) {
          float v = acc[si][sj][r];
          if (bias) v += bias[col];
          if (relu) v = fmaxf(v, 0.0f);
          C[(size_t)(cBase + row) * ldc + col] = v;
        }
      }
}

// ---------------- plain GEMM ----------------
__global__ __launch_bounds__(256)
void gemm_kernel(const float* __restrict__ A, const float* __restrict__ W,
                 const float* __restrict__ bias, float* __restrict__ C,
                 int M, int N, int K, int relu)
{
  gemm_core(A, K, nullptr, 0, W, bias, relu, C, N, 0, M, N, K,
            blockIdx.y * BMt, blockIdx.x * BNt);
}

// ---------------- MoE expert GEMM (device-side M, optional gather, fused relu) ----------------
__global__ __launch_bounds__(256)
void moe_gemm_kernel(const float* __restrict__ A, int lda,
                     const int* __restrict__ rows, int useGather,
                     const float* __restrict__ Wall, const float* __restrict__ ball,
                     float* __restrict__ C, int ldc,
                     const int* __restrict__ offsets, int N, int K, int relu)
{
  const int e    = blockIdx.z;
  const int base = offsets[e];
  const int ne   = offsets[e + 1] - base;
  const int tm0  = blockIdx.y * BMt;
  if (tm0 >= ne) return;                       // uniform early exit (worst-case grid)
  gemm_core(A, lda,
            useGather ? (rows + base) : nullptr, useGather ? 0 : base,
            Wall + (size_t)e * K * N, ball + (size_t)e * N, relu,
            C, ldc, base, ne, N, K, tm0, blockIdx.x * BNt);
}

// ---------------- flash-style causal MLA attention ----------------
// grid: (T/64, NH, B); block 256. q/k/v buffers are [B*T][D] with head h at cols h*64..h*64+63.
// Softmax state is distributed: thread (row = tid&63, quad = tid>>6) owns 16 of the 64 cols.
__global__ __launch_bounds__(256)
void attn_kernel(const float* __restrict__ qb, const float* __restrict__ kb,
                 const float* __restrict__ vb, float* __restrict__ attb)
{
  const int qt = blockIdx.x, h = blockIdx.y, b = blockIdx.z;
  const int tid = threadIdx.x;
  const int lane = tid & 31, wave = tid >> 5;
  const int g = lane >> 4, l16 = lane & 15;
  const int sm = wave & 3, snb = wave >> 2;    // wave -> (m-subtile, n-pair)
  const int rown = tid & 63;                   // owned row
  const int quad = tid >> 6;                   // owned 16-col / 16-d slice

  __shared__ bf16 Qs[64 * 72];
  __shared__ bf16 Ks[64 * 72];
  __shared__ bf16 Vt[64 * 72];                 // V transposed: [d][kv]
  __shared__ bf16 Ps[64 * 72];
  __shared__ float Ss[64 * 65];
  __shared__ float PVs[64 * 65];
  __shared__ float redM[64 * 4];
  __shared__ float redS[64 * 4];

  const int qrow0 = b * Tc + qt * 64;
  const int col0  = h * DKc;

  for (int i = tid; i < 64 * 16; i += 256) {   // float4 chunks
    int r = i >> 4, c = (i & 15) * 4;
    float4 qv = *(const float4*)&qb[(size_t)(qrow0 + r) * Dc + col0 + c];
    *(v4bf*)&Qs[r * 72 + c] = pack4(qv.x, qv.y, qv.z, qv.w);
  }

  float m_i = -3.0e38f, l_i = 0.0f;            // replicated across the 4 quads of a row
  float O[16];
#pragma unroll
  for (int d = 0; d < 16; ++d) O[d] = 0.0f;

  for (int kt = 0; kt <= qt; ++kt) {
    __syncthreads();
    const int krow0 = b * Tc + kt * 64;
    for (int i = tid; i < 64 * 16; i += 256) {
      int r = i >> 4, c = (i & 15) * 4;
      float4 kv = *(const float4*)&kb[(size_t)(krow0 + r) * Dc + col0 + c];
      *(v4bf*)&Ks[r * 72 + c] = pack4(kv.x, kv.y, kv.z, kv.w);
    }
    for (int i = tid; i < 64 * 16; i += 256) { // V transposed: 4 kv's per d
      int d = i & 63, kvb = (i >> 6) * 4;
      float a0 = vb[(size_t)(krow0 + kvb + 0) * Dc + col0 + d];
      float a1 = vb[(size_t)(krow0 + kvb + 1) * Dc + col0 + d];
      float a2 = vb[(size_t)(krow0 + kvb + 2) * Dc + col0 + d];
      float a3 = vb[(size_t)(krow0 + kvb + 3) * Dc + col0 + d];
      *(v4bf*)&Vt[d * 72 + kvb] = pack4(a0, a1, a2, a3);
    }
    __syncthreads();

    // S = (Q @ K^T) * 1/sqrt(DK), causal mask
    {
      int mrow = sm * 16 + l16;
#pragma unroll
      for (int j = 0; j < 2; ++j) {
        int ncol = (snb * 2 + j) * 16 + l16;
        v8f acc;
#pragma unroll
        for (int r = 0; r < 8; ++r) acc[r] = 0.0f;
#pragma unroll
        for (int ks = 0; ks < 2; ++ks) {
          v16bf a  = frag_a(Qs, mrow, 72, ks * 32, g);
          v16bf bb = frag_b(Ks, ncol, 72, ks * 32, g);
          acc = __builtin_amdgcn_wmma_f32_16x16x32_bf16(false, a, false, bb, (short)0, acc, false, false);
        }
#pragma unroll
        for (int r = 0; r < 8; ++r) {
          int mm = sm * 16 + g * 8 + r;
          int nn = (snb * 2 + j) * 16 + l16;
          float s = acc[r] * 0.125f;
          if (kt * 64 + nn > qt * 64 + mm) s = NEGF;
          Ss[mm * 65 + nn] = s;
        }
      }
    }
    __syncthreads();

    // distributed online softmax: 4 threads per row
    float pm = -3.0e38f;
#pragma unroll
    for (int j = 0; j < 16; ++j) pm = fmaxf(pm, Ss[rown * 65 + quad * 16 + j]);
    redM[rown * 4 + quad] = pm;
    __syncthreads();
    float mx = fmaxf(m_i, fmaxf(fmaxf(redM[rown * 4 + 0], redM[rown * 4 + 1]),
                                fmaxf(redM[rown * 4 + 2], redM[rown * 4 + 3])));
    float ps = 0.0f;
#pragma unroll
    for (int j = 0; j < 16; j += 2) {
      float p0 = __expf(Ss[rown * 65 + quad * 16 + j] - mx);
      float p1 = __expf(Ss[rown * 65 + quad * 16 + j + 1] - mx);
      v2bf pp; pp[0] = (bf16)p0; pp[1] = (bf16)p1;
      *(v2bf*)&Ps[rown * 72 + quad * 16 + j] = pp;
      ps += p0 + p1;
    }
    redS[rown * 4 + quad] = ps;
    __syncthreads();                            // Ps + redS visible to all
    {
      float ls = redS[rown * 4 + 0] + redS[rown * 4 + 1]
               + redS[rown * 4 + 2] + redS[rown * 4 + 3];
      float scl = __expf(m_i - mx);
      l_i = l_i * scl + ls;
      m_i = mx;
#pragma unroll
      for (int d = 0; d < 16; ++d) O[d] *= scl;
    }

    // PV = P @ V
    {
      int mrow = sm * 16 + l16;
#pragma unroll
      for (int j = 0; j < 2; ++j) {
        int ncol = (snb * 2 + j) * 16 + l16;
        v8f acc;
#pragma unroll
        for (int r = 0; r < 8; ++r) acc[r] = 0.0f;
#pragma unroll
        for (int ks = 0; ks < 2; ++ks) {
          v16bf a  = frag_a(Ps, mrow, 72, ks * 32, g);
          v16bf bb = frag_b(Vt, ncol, 72, ks * 32, g);
          acc = __builtin_amdgcn_wmma_f32_16x16x32_bf16(false, a, false, bb, (short)0, acc, false, false);
        }
#pragma unroll
        for (int r = 0; r < 8; ++r)
          PVs[(sm * 16 + g * 8 + r) * 65 + ncol] = acc[r];
      }
    }
    __syncthreads();
#pragma unroll
    for (int d = 0; d < 16; ++d) O[d] += PVs[rown * 65 + quad * 16 + d];
  }

  float inv = 1.0f / l_i;
#pragma unroll
  for (int d = 0; d < 16; d += 4) {
    float4 o4 = make_float4(O[d] * inv, O[d + 1] * inv, O[d + 2] * inv, O[d + 3] * inv);
    *(float4*)&attb[(size_t)(qrow0 + rown) * Dc + col0 + quad * 16 + d] = o4;
  }
}

// ---------------- residual add + LayerNorm (one row per block, float4) ----------------
__global__ __launch_bounds__(256)
void add_ln_kernel(const float* __restrict__ xa, const float* __restrict__ xb,
                   const float* __restrict__ gamma, const float* __restrict__ beta,
                   float* __restrict__ out)
{
  const int row = blockIdx.x, tid = threadIdx.x;
  __shared__ float sb[256], sb2[256];
  float4 va = *(const float4*)&xa[(size_t)row * Dc + tid * 4];
  float4 vb2 = *(const float4*)&xb[(size_t)row * Dc + tid * 4];
  float v[4] = { va.x + vb2.x, va.y + vb2.y, va.z + vb2.z, va.w + vb2.w };
  float s = 0.0f, s2 = 0.0f;
#pragma unroll
  for (int i = 0; i < 4; ++i) { s += v[i]; s2 += v[i] * v[i]; }
  sb[tid] = s; sb2[tid] = s2; __syncthreads();
  for (int off = 128; off > 0; off >>= 1) {
    if (tid < off) { sb[tid] += sb[tid + off]; sb2[tid] += sb2[tid + off]; }
    __syncthreads();
  }
  float mean = sb[0] * (1.0f / Dc);
  float var  = sb2[0] * (1.0f / Dc) - mean * mean;
  float rstd = rsqrtf(var + 1e-5f);
  float4 g4 = *(const float4*)&gamma[tid * 4];
  float4 b4 = *(const float4*)&beta[tid * 4];
  float4 o4 = make_float4((v[0] - mean) * rstd * g4.x + b4.x,
                          (v[1] - mean) * rstd * g4.y + b4.y,
                          (v[2] - mean) * rstd * g4.z + b4.z,
                          (v[3] - mean) * rstd * g4.w + b4.w);
  *(float4*)&out[(size_t)row * Dc + tid * 4] = o4;
}

// ---------------- JAX threefry2x32 noise (key = (0, 42)) ----------------
__device__ __forceinline__ unsigned rotl32(unsigned x, int r) { return (x << r) | (x >> (32 - r)); }

__device__ __forceinline__ void threefry2x32(unsigned k0, unsigned k1,
                                             unsigned x0, unsigned x1,
                                             unsigned* o0, unsigned* o1)
{
  unsigned ks[3] = { k0, k1, 0x1BD11BDAu ^ k0 ^ k1 };
  const int R0[4] = {13, 15, 26, 6};
  const int R1[4] = {17, 29, 16, 24};
  x0 += ks[0]; x1 += ks[1];
#pragma unroll
  for (int grp = 0; grp < 5; ++grp) {
    const int* rr = (grp & 1) ? R1 : R0;
#pragma unroll
    for (int i = 0; i < 4; ++i) { x0 += x1; x1 = rotl32(x1, rr[i]); x1 ^= x0; }
    x0 += ks[(grp + 1) % 3];
    x1 += ks[(grp + 2) % 3] + (unsigned)(grp + 1);
  }
  *o0 = x0; *o1 = x1;
}

__device__ __forceinline__ float erfinvf_(float x) {
  float w = -__logf((1.0f - x) * (1.0f + x));
  float p;
  if (w < 5.0f) {
    w -= 2.5f;
    p = 2.81022636e-08f;
    p = 3.43273939e-07f + p * w;
    p = -3.5233877e-06f + p * w;
    p = -4.39150654e-06f + p * w;
    p = 0.00021858087f + p * w;
    p = -0.00125372503f + p * w;
    p = -0.00417768164f + p * w;
    p = 0.246640727f + p * w;
    p = 1.50140941f + p * w;
  } else {
    w = sqrtf(w) - 3.0f;
    p = -0.000200214257f;
    p = 0.000100950558f + p * w;
    p = 0.00134934322f + p * w;
    p = -0.00367342844f + p * w;
    p = 0.00573950773f + p * w;
    p = -0.0076224613f + p * w;
    p = 0.00943887047f + p * w;
    p = 1.00167406f + p * w;
    p = 2.83297682f + p * w;
  }
  return p * x;
}

__device__ __forceinline__ float jax_noise(int flat) {
  const int Ntot = Mtok * Ec;            // 32768, even
  const int half = Ntot / 2;
  unsigned o0, o1, bits;
  if (flat < half) { threefry2x32(0u, 42u, (unsigned)flat, (unsigned)(flat + half), &o0, &o1); bits = o0; }
  else             { threefry2x32(0u, 42u, (unsigned)(flat - half), (unsigned)flat, &o0, &o1); bits = o1; }
  float f = __uint_as_float((bits >> 9) | 0x3f800000u) - 1.0f;   // [0,1)
  const float lo = -0.99999994f;                                  // nextafter(-1,0)
  float u = f * (1.0f - lo) + lo;                                 // [-1+eps, 1)
  return 1.41421356f * erfinvf_(u);
}

// ---------------- gating: one wave per token ----------------
__global__ __launch_bounds__(256)
void gate_kernel(const float* __restrict__ x1, const float* __restrict__ Wg,
                 const float* __restrict__ bg, const float* __restrict__ Wn,
                 const float* __restrict__ bn,
                 int* __restrict__ topIdx, float* __restrict__ topSc,
                 int* __restrict__ counts)
{
  const int token = blockIdx.x * 8 + (threadIdx.x >> 5);
  const int lane  = threadIdx.x & 31;
  if (token >= Mtok) return;
  float ag[Ec], an[Ec];
#pragma unroll
  for (int e = 0; e < Ec; ++e) { ag[e] = 0.0f; an[e] = 0.0f; }
  for (int d = lane; d < Dc; d += 32) {
    float xv = x1[(size_t)token * Dc + d];
    const float4* wg4 = (const float4*)(Wg + (size_t)d * Ec);
    const float4* wn4 = (const float4*)(Wn + (size_t)d * Ec);
    float4 g0 = wg4[0], g1 = wg4[1];
    float4 n0 = wn4[0], n1 = wn4[1];
    ag[0] += xv * g0.x; ag[1] += xv * g0.y; ag[2] += xv * g0.z; ag[3] += xv * g0.w;
    ag[4] += xv * g1.x; ag[5] += xv * g1.y; ag[6] += xv * g1.z; ag[7] += xv * g1.w;
    an[0] += xv * n0.x; an[1] += xv * n0.y; an[2] += xv * n0.z; an[3] += xv * n0.w;
    an[4] += xv * n1.x; an[5] += xv * n1.y; an[6] += xv * n1.z; an[7] += xv * n1.w;
  }
#pragma unroll
  for (int e = 0; e < Ec; ++e)
    for (int off = 16; off > 0; off >>= 1) {
      ag[e] += __shfl_xor(ag[e], off, 32);
      an[e] += __shfl_xor(an[e], off, 32);
    }
  if (lane == 0) {
    float H[Ec];
#pragma unroll
    for (int e = 0; e < Ec; ++e) {
      float gg = ag[e] + bg[e];
      float nn = an[e] + bn[e];
      float sp = fmaxf(nn, 0.0f) + log1pf(__expf(-fabsf(nn)));    // stable softplus
      H[e] = gg + jax_noise(token * Ec + e) * sp;
    }
    int e0 = 0;
    for (int e = 1; e < Ec; ++e) if (H[e] > H[e0]) e0 = e;
    int e1 = (e0 == 0) ? 1 : 0;
    for (int e = 0; e < Ec; ++e) if (e != e0 && H[e] > H[e1]) e1 = e;
    float a1 = __expf(H[e1] - H[e0]);
    float inv = 1.0f / (1.0f + a1);
    topIdx[token * 2 + 0] = e0; topIdx[token * 2 + 1] = e1;
    topSc[token * 2 + 0]  = inv; topSc[token * 2 + 1] = a1 * inv;
    atomicAdd(&counts[e0], 1);
    atomicAdd(&counts[e1], 1);
  }
}

__global__ void zero_int_kernel(int* __restrict__ p, int n) {
  int i = blockIdx.x * blockDim.x + threadIdx.x;
  if (i < n) p[i] = 0;
}

__global__ void scan_kernel(const int* __restrict__ counts, int* __restrict__ offsets,
                            int* __restrict__ cursor) {
  if (threadIdx.x == 0) {
    int acc = 0;
    for (int e = 0; e < Ec; ++e) { offsets[e] = acc; acc += counts[e]; cursor[e] = 0; }
    offsets[Ec] = acc;   // == NSLOT
  }
}

__global__ void fill_kernel(const int* __restrict__ topIdx, const int* __restrict__ offsets,
                            int* __restrict__ cursor, int* __restrict__ rows,
                            int* __restrict__ slotOf) {
  int t = blockIdx.x * blockDim.x + threadIdx.x;
  if (t >= Mtok) return;
  for (int k2 = 0; k2 < Kc; ++k2) {
    int e = topIdx[t * 2 + k2];
    int pos = atomicAdd(&cursor[e], 1);
    int slot = offsets[e] + pos;
    rows[slot] = t;
    slotOf[t * 2 + k2] = slot;
  }
}

// ---------------- combine top-2 experts + residual + final LayerNorm (float4) ----------------
__global__ __launch_bounds__(256)
void moe_ln_kernel(const float* __restrict__ x1, const float* __restrict__ Outg,
                   const int* __restrict__ slotOf, const float* __restrict__ topSc,
                   const float* __restrict__ gamma, const float* __restrict__ beta,
                   float* __restrict__ out)
{
  const int row = blockIdx.x, tid = threadIdx.x;
  const int s0 = slotOf[row * 2 + 0], s1 = slotOf[row * 2 + 1];
  const float w0 = topSc[row * 2 + 0], w1 = topSc[row * 2 + 1];
  __shared__ float sb[256], sb2[256];
  float4 vx = *(const float4*)&x1[(size_t)row * Dc + tid * 4];
  float4 v0 = *(const float4*)&Outg[(size_t)s0 * Dc + tid * 4];
  float4 v1 = *(const float4*)&Outg[(size_t)s1 * Dc + tid * 4];
  float v[4] = { vx.x + w0 * v0.x + w1 * v1.x,
                 vx.y + w0 * v0.y + w1 * v1.y,
                 vx.z + w0 * v0.z + w1 * v1.z,
                 vx.w + w0 * v0.w + w1 * v1.w };
  float s = 0.0f, s2 = 0.0f;
#pragma unroll
  for (int i = 0; i < 4; ++i) { s += v[i]; s2 += v[i] * v[i]; }
  sb[tid] = s; sb2[tid] = s2; __syncthreads();
  for (int off = 128; off > 0; off >>= 1) {
    if (tid < off) { sb[tid] += sb[tid + off]; sb2[tid] += sb2[tid + off]; }
    __syncthreads();
  }
  float mean = sb[0] * (1.0f / Dc);
  float var  = sb2[0] * (1.0f / Dc) - mean * mean;
  float rstd = rsqrtf(var + 1e-5f);
  float4 g4 = *(const float4*)&gamma[tid * 4];
  float4 b4 = *(const float4*)&beta[tid * 4];
  float4 o4 = make_float4((v[0] - mean) * rstd * g4.x + b4.x,
                          (v[1] - mean) * rstd * g4.y + b4.y,
                          (v[2] - mean) * rstd * g4.z + b4.z,
                          (v[3] - mean) * rstd * g4.w + b4.w);
  *(float4*)&out[(size_t)row * Dc + tid * 4] = o4;
}

// ---------------- host launch ----------------
extern "C" void kernel_launch(void* const* d_in, const int* in_sizes, int n_in,
                              void* d_out, int out_size, void* d_ws, size_t ws_size,
                              hipStream_t stream) {
  const float* x     = (const float*)d_in[0];
  const float* Wdkv  = (const float*)d_in[1];
  const float* bdkv  = (const float*)d_in[2];
  const float* Wuk   = (const float*)d_in[3];
  const float* buk   = (const float*)d_in[4];
  const float* Wuv   = (const float*)d_in[5];
  const float* buv   = (const float*)d_in[6];
  const float* Wdq   = (const float*)d_in[7];
  const float* bdq   = (const float*)d_in[8];
  const float* Wuq   = (const float*)d_in[9];
  const float* buq   = (const float*)d_in[10];
  const float* Wo    = (const float*)d_in[11];
  const float* bo    = (const float*)d_in[12];
  const float* gamma1= (const float*)d_in[13];
  const float* beta1 = (const float*)d_in[14];
  const float* gamma2= (const float*)d_in[15];
  const float* beta2 = (const float*)d_in[16];
  const float* Wg    = (const float*)d_in[17];
  const float* bg    = (const float*)d_in[18];
  const float* Wn    = (const float*)d_in[19];
  const float* bn    = (const float*)d_in[20];
  const float* W1    = (const float*)d_in[21];
  const float* b1    = (const float*)d_in[22];
  const float* W2    = (const float*)d_in[23];
  const float* b2    = (const float*)d_in[24];

  char* wsb = (char*)d_ws;
  size_t cur = 0;
  auto carve = [&](size_t bytes) -> void* {
    void* p = wsb + cur;
    cur += (bytes + 255) & ~(size_t)255;
    return p;
  };

  float* ckv  = (float*)carve((size_t)Mtok * Lc * 4);
  float* cq   = (float*)carve((size_t)Mtok * Lc * 4);
  float* kb   = (float*)carve((size_t)Mtok * Dc * 4);
  float* vb   = (float*)carve((size_t)Mtok * Dc * 4);
  float* qb   = (float*)carve((size_t)Mtok * Dc * 4);
  float* attb = (float*)carve((size_t)Mtok * Dc * 4);
  float* attp = (float*)carve((size_t)Mtok * Dc * 4);
  float* x1   = (float*)carve((size_t)Mtok * Dc * 4);
  int*   tIdx = (int*)  carve((size_t)Mtok * Kc * 4);
  float* tSc  = (float*)carve((size_t)Mtok * Kc * 4);
  int*   counts  = (int*)carve(Ec * 4);
  int*   offsets = (int*)carve((Ec + 1) * 4);
  int*   cursor  = (int*)carve(Ec * 4);
  int*   rows    = (int*)carve((size_t)NSLOT * 4);
  int*   slotOf  = (int*)carve((size_t)NSLOT * 4);
  float* Hbuf = (float*)carve((size_t)NSLOT * FFc * 4);   // 128 MB
  float* Outg = (float*)carve((size_t)NSLOT * Dc * 4);    // 32 MB

  dim3 blk(256);
  auto gemm = [&](const float* A, const float* W, const float* bias, float* C,
                  int M_, int N_, int K_, int relu) {
    dim3 grid((N_ + BNt - 1) / BNt, (M_ + BMt - 1) / BMt);
    gemm_kernel<<<grid, blk, 0, stream>>>(A, W, bias, C, M_, N_, K_, relu);
  };

  // ---- MLA projections ----
  gemm(x,   Wdkv, bdkv, ckv, Mtok, Lc, Dc, 0);
  gemm(x,   Wdq,  bdq,  cq,  Mtok, Lc, Dc, 0);
  gemm(ckv, Wuk,  buk,  kb,  Mtok, Dc, Lc, 0);
  gemm(ckv, Wuv,  buv,  vb,  Mtok, Dc, Lc, 0);
  gemm(cq,  Wuq,  buq,  qb,  Mtok, Dc, Lc, 0);

  // ---- attention ----
  attn_kernel<<<dim3(Tc / 64, NHc, Bc), blk, 0, stream>>>(qb, kb, vb, attb);
  gemm(attb, Wo, bo, attp, Mtok, Dc, Dc, 0);

  // ---- x1 = LN(x + att) ----
  add_ln_kernel<<<Mtok, blk, 0, stream>>>(x, attp, gamma1, beta1, x1);

  // ---- gating + routing ----
  zero_int_kernel<<<1, 32, 0, stream>>>(counts, Ec);
  gate_kernel<<<Mtok / 8, blk, 0, stream>>>(x1, Wg, bg, Wn, bn, tIdx, tSc, counts);
  scan_kernel<<<1, 32, 0, stream>>>(counts, offsets, cursor);
  fill_kernel<<<(Mtok + 255) / 256, blk, 0, stream>>>(tIdx, offsets, cursor, rows, slotOf);

  // ---- expert FFN (only routed tokens; worst-case grid, device-side M) ----
  moe_gemm_kernel<<<dim3(FFc / BNt, (Mtok + BMt - 1) / BMt, Ec), blk, 0, stream>>>(
      x1, Dc, rows, 1, W1, b1, Hbuf, FFc, offsets, FFc, Dc, 1);
  moe_gemm_kernel<<<dim3(Dc / BNt, (Mtok + BMt - 1) / BMt, Ec), blk, 0, stream>>>(
      Hbuf, FFc, nullptr, 0, W2, b2, Outg, Dc, offsets, Dc, FFc, 0);

  // ---- out = LN(x1 + sum_k sc_k * expert_out_k) ----
  moe_ln_kernel<<<Mtok, blk, 0, stream>>>(x1, Outg, slotOf, tSc, gamma2, beta2,
                                          (float*)d_out);
}
